// DABlock_58428735095193
// MI455X (gfx1250) — compile-verified, matrix operand order checked by reading, not run on hardware
//
#include <hip/hip_runtime.h>
#include <hip/hip_bf16.h>

typedef _Float16 h16;
typedef __attribute__((ext_vector_type(16))) _Float16 v16h;
typedef __attribute__((ext_vector_type(8)))  _Float16 v8h;
typedef __attribute__((ext_vector_type(8)))  float    v8f;
typedef int v4si __attribute__((__vector_size__(16)));   // matches builtin param type

#define DEV __device__ __forceinline__

// Model dims
#define BB    16
#define CC    384
#define HW_   3136
#define NH_   12
#define HD_   32
#define MLP_  1536
#define NTOK  50176L          // BB*HW_
#define BCN   19267584L       // BB*CC*HW_

#if __has_builtin(__builtin_amdgcn_global_load_async_to_lds_b128) && \
    __has_builtin(__builtin_amdgcn_s_wait_asynccnt)
#define HAVE_ASYNC_LDS 1
#else
#define HAVE_ASYNC_LDS 0
#endif

union Frag { v16h v; v8h h[2]; };

DEV v8f zero8(){ v8f z; for (int i = 0; i < 8; ++i) z[i] = 0.f; return z; }
DEV float wsum(float v){
  v += __shfl_xor(v,16); v += __shfl_xor(v,8); v += __shfl_xor(v,4);
  v += __shfl_xor(v,2);  v += __shfl_xor(v,1); return v;
}
DEV float wminf(float v){
  v = fminf(v,__shfl_xor(v,16)); v = fminf(v,__shfl_xor(v,8)); v = fminf(v,__shfl_xor(v,4));
  v = fminf(v,__shfl_xor(v,2));  v = fminf(v,__shfl_xor(v,1)); return v;
}
DEV float geluf(float x){
  return 0.5f*x*(1.f + tanhf(0.7978845608028654f*(x + 0.044715f*x*x*x)));
}
// Fragment load; per ISA: lane<16 -> K = {0..7,16..23}; lane>=16 -> +8.
DEV Frag ldfrag(const h16* p, int kb){
  Frag f; f.h[0] = *(const v8h*)(p + kb); f.h[1] = *(const v8h*)(p + kb + 16); return f;
}

// 16-byte global -> LDS copy: async on gfx1250 (ASYNCcnt), sync fallback.
DEV void cp16(const h16* g, h16* l){
#if HAVE_ASYNC_LDS
  __builtin_amdgcn_global_load_async_to_lds_b128(
      (__attribute__((address_space(1))) v4si*)g,
      (__attribute__((address_space(3))) v4si*)l, 0, 0);
#else
  *(v8h*)l = *(const v8h*)g;
#endif
}
DEV void wait_async_3(){
#if HAVE_ASYNC_LDS
  __builtin_amdgcn_s_wait_asynccnt(3);
#endif
}
DEV void wait_async_0(){
#if HAVE_ASYNC_LDS
  __builtin_amdgcn_s_wait_asynccnt(0);
#endif
}

// ---------------- 128x64 double-buffered async WMMA GEMM mainloop -----------
// A row-major [M,K]; B row-major [N,K] (weight style).  Block = 256 thr = 8
// waves; wave(wid) owns rows [row0+wid*16, +16) x all 64 cols (4 accum tiles).
DEV void gemm_mainloop2(const h16* __restrict__ A, const h16* __restrict__ Bm,
                        int K, int lda, int ldb, int row0, int col0,
                        v8f (&acc)[4]){
  __shared__ h16 As[2][128*32];
  __shared__ h16 Bs[2][64*32];
  const int tid = threadIdx.x, lane = tid & 31, wid = tid >> 5;
  const int kb = (lane & 16) ? 8 : 0, lr = lane & 15;
  for (int i = 0; i < 4; ++i) acc[i] = zero8();
  const int ar = tid >> 2, akk = (tid & 3)*8;   // 64 rows x 4 chunks per issue
  auto stage = [&](int buf, int k0){
    cp16(&A[(long)(row0 + ar)*lda + k0 + akk],      &As[buf][ar*32 + akk]);
    cp16(&A[(long)(row0 + ar + 64)*lda + k0 + akk], &As[buf][(ar + 64)*32 + akk]);
    cp16(&Bm[(long)(col0 + ar)*ldb + k0 + akk],     &Bs[buf][ar*32 + akk]);
  };
  stage(0, 0);
  for (int k0 = 0; k0 < K; k0 += 32){
    int buf = (k0 >> 5) & 1;
    if (k0 + 32 < K){ stage(buf ^ 1, k0 + 32); wait_async_3(); }
    else            { wait_async_0(); }
    __syncthreads();
    Frag af = ldfrag(&As[buf][(wid*16 + lr)*32], kb);
    #pragma unroll
    for (int tc = 0; tc < 4; ++tc){
      Frag bf = ldfrag(&Bs[buf][(tc*16 + lr)*32], kb);
      acc[tc] = __builtin_amdgcn_wmma_f32_16x16x32_f16(false, af.v, false, bf.v,
                                                       (short)0, acc[tc], false, false);
    }
    __syncthreads();
  }
}

// ---------------- 64x64 sync mainloop, B row-major [K,N] (camout only) ------
DEV void gemm_mainloop_kn(const h16* __restrict__ A, const h16* __restrict__ Bm,
                          int K, int lda, int ldb, int row0, int col0,
                          v8f& acc0, v8f& acc1){
  __shared__ h16 As[64*32];
  __shared__ h16 Bs[64*32];
  const int tid = threadIdx.x, lane = tid & 31, wid = tid >> 5;
  const int wr = (wid & 3)*16, wc = (wid >> 2)*32;
  const int kb = (lane & 16) ? 8 : 0, lr = lane & 15;
  acc0 = zero8(); acc1 = zero8();
  for (int k0 = 0; k0 < K; k0 += 32){
    { int r = tid >> 2, kk = (tid & 3)*8;
      *(v8h*)&As[r*32 + kk] = *(const v8h*)&A[(long)(row0 + r)*lda + k0 + kk]; }
    for (int j = 0; j < 8; ++j){
      int idx = j*256 + tid;
      int kk = idx >> 6, c = idx & 63;           // coalesced over c
      Bs[c*32 + kk] = Bm[(long)(k0 + kk)*ldb + col0 + c];
    }
    __syncthreads();
    Frag af = ldfrag(&As[(wr + lr)*32], kb);
    Frag b0 = ldfrag(&Bs[(wc + lr)*32], kb);
    Frag b1 = ldfrag(&Bs[(wc + 16 + lr)*32], kb);
    acc0 = __builtin_amdgcn_wmma_f32_16x16x32_f16(false, af.v, false, b0.v, (short)0, acc0, false, false);
    acc1 = __builtin_amdgcn_wmma_f32_16x16x32_f16(false, af.v, false, b1.v, (short)0, acc1, false, false);
    __syncthreads();
  }
}

// ---------------- casts / reorders ------------------------------------------
__global__ void k_cast(const float* __restrict__ in, h16* __restrict__ o, long n){
  long i = (long)blockIdx.x*blockDim.x + threadIdx.x;
  long s = (long)gridDim.x*blockDim.x;
  for (; i < n; i += s) o[i] = (h16)in[i];
}
// conv w [co][ci][ky][kx] f32 -> [co][(ky*3+kx)*CI + ci] f16
__global__ void k_convw(const float* __restrict__ w, h16* __restrict__ o, int CI){
  long n = (long)CC*CI*9;
  long i = (long)blockIdx.x*blockDim.x + threadIdx.x;
  long s = (long)gridDim.x*blockDim.x;
  for (; i < n; i += s){
    int r  = (int)(i % 9);
    long t = i / 9;
    int ci = (int)(t % CI);
    int co = (int)(t / CI);
    o[(long)co*CI*9 + (long)r*CI + ci] = (h16)w[i];
  }
}
// [B,C,HW] -> [B,HW,C]
__global__ void k_transpose(const float* __restrict__ in, float* __restrict__ o){
  long i = (long)blockIdx.x*blockDim.x + threadIdx.x;
  long s = (long)gridDim.x*blockDim.x;
  for (; i < BCN; i += s){
    int hw = (int)(i % HW_);
    long r = i / HW_;
    int c  = (int)(r % CC);
    int b  = (int)(r / CC);
    o[((long)b*HW_ + hw)*CC + c] = in[i];
  }
}

// ---------------- implicit-GEMM 3x3 SAME conv + BN + GELU -------------------
// K order: k = (ky*3+kx)*CI + ci  (tap-major), so each 32-wide K-step is one
// tap and 32 consecutive input channels.  Per-thread im2col coords hoisted.
__global__ __launch_bounds__(256) void k_conv3x3(
    const h16* __restrict__ xin, const h16* __restrict__ W,
    const float* __restrict__ scale, const float* __restrict__ bias,
    float* __restrict__ outf, h16* __restrict__ outh, int CI){
  __shared__ h16 As[64*32];
  __shared__ h16 Bs[64*32];
  const int tid = threadIdx.x, lane = tid & 31, wid = tid >> 5;
  const int wr = (wid & 3)*16, wc = (wid >> 2)*32;
  const int row0 = blockIdx.y*64, col0 = blockIdx.x*64;
  const int K = 9*CI;
  const int kb = (lane & 16) ? 8 : 0, lr = lane & 15;
  // K-invariant im2col coordinates for this thread's staging row
  const int mi = tid & 63;             // As row this thread fills
  const int cb = tid >> 6;             // channel sub-slot (0..3)
  const int m  = row0 + mi;
  const int bimg = m / HW_, hw = m % HW_;
  const int hh0 = hw / 56, ww0 = hw % 56;
  const long srcb = (long)bimg*CI*HW_;
  v8f acc0 = zero8(), acc1 = zero8();
  for (int r = 0; r < 9; ++r){
    const int dy = r/3 - 1, dx = r%3 - 1;
    const int hh = hh0 + dy, ww = ww0 + dx;
    const bool ok = ((unsigned)hh < 56u) && ((unsigned)ww < 56u);
    const h16* src = xin + srcb + hh*56 + ww;
    for (int ci0 = 0; ci0 < CI; ci0 += 32){
      #pragma unroll
      for (int j = 0; j < 8; ++j){
        int cii = j*4 + cb;
        As[mi*32 + cii] = ok ? src[(long)(ci0 + cii)*HW_] : (h16)0.f;
      }
      { int c = tid >> 2, kk = (tid & 3)*8;
        *(v8h*)&Bs[c*32 + kk] = *(const v8h*)&W[(long)(col0 + c)*K + r*CI + ci0 + kk]; }
      __syncthreads();
      Frag af = ldfrag(&As[(wr + lr)*32], kb);
      Frag b0 = ldfrag(&Bs[(wc + lr)*32], kb);
      Frag b1 = ldfrag(&Bs[(wc + 16 + lr)*32], kb);
      acc0 = __builtin_amdgcn_wmma_f32_16x16x32_f16(false, af.v, false, b0.v, (short)0, acc0, false, false);
      acc1 = __builtin_amdgcn_wmma_f32_16x16x32_f16(false, af.v, false, b1.v, (short)0, acc1, false, false);
      __syncthreads();
    }
  }
  const int rb = (lane & 16) ? 8 : 0, lc = lane & 15;
  for (int hf = 0; hf < 2; ++hf){
    v8f& a = hf ? acc1 : acc0;
    int co = col0 + wc + hf*16 + lc;
    float sc = scale[co], bi = bias[co];
    for (int rr = 0; rr < 8; ++rr){
      int mm = row0 + wr + rb + rr;
      int b = mm / HW_, hw2 = mm % HW_;
      float y = geluf(a[rr]*sc + bi);
      long oi = ((long)b*CC + co)*HW_ + hw2;
      outf[oi] = y;
      if (outh) outh[oi] = (h16)y;
    }
  }
}

// ---------------- GEMM kernels with custom epilogues ------------------------
__global__ __launch_bounds__(256) void k_gemm_qkv(
    const h16* __restrict__ A, const h16* __restrict__ W, const float* __restrict__ bias,
    h16* __restrict__ out, int N, int K){
  v8f acc[4];
  gemm_mainloop2(A, W, K, K, K, blockIdx.y*128, blockIdx.x*64, acc);
  const int lane = threadIdx.x & 31, wid = threadIdx.x >> 5;
  const int rb = (lane & 16) ? 8 : 0, lc = lane & 15;
  const long rowbase = (long)blockIdx.y*128 + wid*16 + rb;
  for (int tc = 0; tc < 4; ++tc){
    int col = blockIdx.x*64 + tc*16 + lc;
    float bv = bias[col];
    for (int r = 0; r < 8; ++r)
      out[(rowbase + r)*N + col] = (h16)(acc[tc][r] + bv);
  }
}

__global__ __launch_bounds__(256) void k_gemm_proj(
    const h16* __restrict__ A, const h16* __restrict__ W, const float* __restrict__ bias,
    float* __restrict__ t, int K, int shift){
  v8f acc[4];
  gemm_mainloop2(A, W, K, K, K, blockIdx.y*128, blockIdx.x*64, acc);
  const int lane = threadIdx.x & 31, wid = threadIdx.x >> 5;
  const int rb = (lane & 16) ? 8 : 0, lc = lane & 15;
  const int rowbase = blockIdx.y*128 + wid*16 + rb;
  for (int tc = 0; tc < 4; ++tc){
    int col = blockIdx.x*64 + tc*16 + lc;
    float bv = bias[col];
    for (int r = 0; r < 8; ++r){
      int m = rowbase + r;
      int win = m/49, pos = m%49;
      int b = win >> 6, wloc = win & 63;
      int h2 = (wloc >> 3)*7 + pos/7, w2 = (wloc & 7)*7 + pos%7;
      int hh = h2 + shift; if (hh >= 56) hh -= 56;
      int ww = w2 + shift; if (ww >= 56) ww -= 56;
      long ti = ((long)b*HW_ + hh*56 + ww)*CC + col;
      t[ti] += acc[tc][r] + bv;                  // shortcut + proj output
    }
  }
}

__global__ __launch_bounds__(256) void k_gemm_fc1(
    const h16* __restrict__ A, const h16* __restrict__ W, const float* __restrict__ bias,
    h16* __restrict__ out, int N, int K){
  v8f acc[4];
  gemm_mainloop2(A, W, K, K, K, blockIdx.y*128, blockIdx.x*64, acc);
  const int lane = threadIdx.x & 31, wid = threadIdx.x >> 5;
  const int rb = (lane & 16) ? 8 : 0, lc = lane & 15;
  const long rowbase = (long)blockIdx.y*128 + wid*16 + rb;
  for (int tc = 0; tc < 4; ++tc){
    int col = blockIdx.x*64 + tc*16 + lc;
    float bv = bias[col];
    for (int r = 0; r < 8; ++r)
      out[(rowbase + r)*N + col] = (h16)geluf(acc[tc][r] + bv);
  }
}

__global__ __launch_bounds__(256) void k_gemm_fc2(
    const h16* __restrict__ A, const h16* __restrict__ W, const float* __restrict__ bias,
    float* __restrict__ t, int K){
  v8f acc[4];
  gemm_mainloop2(A, W, K, K, K, blockIdx.y*128, blockIdx.x*64, acc);
  const int lane = threadIdx.x & 31, wid = threadIdx.x >> 5;
  const int rb = (lane & 16) ? 8 : 0, lc = lane & 15;
  const long rowbase = (long)blockIdx.y*128 + wid*16 + rb;
  for (int tc = 0; tc < 4; ++tc){
    int col = blockIdx.x*64 + tc*16 + lc;
    float bv = bias[col];
    for (int r = 0; r < 8; ++r)
      t[(rowbase + r)*CC + col] += acc[tc][r] + bv;   // residual
  }
}

__global__ __launch_bounds__(256) void k_gemm_energy(const h16* __restrict__ X,
                                                     float* __restrict__ E){
  long bo = (long)blockIdx.z*CC*HW_;
  v8f acc[4];
  gemm_mainloop2(X + bo, X + bo, HW_, HW_, HW_, blockIdx.y*128, blockIdx.x*64, acc);
  const int lane = threadIdx.x & 31, wid = threadIdx.x >> 5;
  const int rb = (lane & 16) ? 8 : 0, lc = lane & 15;
  const int rowbase = blockIdx.y*128 + wid*16 + rb;
  for (int tc = 0; tc < 4; ++tc){
    int col = blockIdx.x*64 + tc*16 + lc;
    for (int r = 0; r < 8; ++r)
      E[((long)blockIdx.z*CC + rowbase + r)*CC + col] = acc[tc][r];
  }
}

__global__ __launch_bounds__(256) void k_gemm_camout(
    const h16* __restrict__ attn, const h16* __restrict__ X, float* __restrict__ O){
  const h16* Ab = attn + (long)blockIdx.z*CC*CC;
  const h16* Bb = X    + (long)blockIdx.z*CC*HW_;
  v8f a0, a1;
  gemm_mainloop_kn(Ab, Bb, CC, CC, HW_, blockIdx.y*64, blockIdx.x*64, a0, a1);
  const int lane = threadIdx.x & 31, wid = threadIdx.x >> 5;
  const int wr = (wid & 3)*16, wc = (wid >> 2)*32;
  const int rb = (lane & 16) ? 8 : 0, lc = lane & 15;
  for (int hf = 0; hf < 2; ++hf){
    v8f& a = hf ? a1 : a0;
    int col = blockIdx.x*64 + wc + hf*16 + lc;
    for (int r = 0; r < 8; ++r){
      int row = blockIdx.y*64 + wr + rb + r;
      O[((long)blockIdx.z*CC + row)*HW_ + col] = a[r];
    }
  }
}

// ---------------- attention per (window, head) ------------------------------
__global__ __launch_bounds__(256) void k_attn(
    const h16* __restrict__ qkv, const float* __restrict__ rpb,
    h16* __restrict__ outp, int shifted){
  __shared__ h16 Qs[64*32];
  __shared__ h16 Ks[64*32];
  __shared__ h16 Vt[32*64];     // V transposed: [d][j]
  __shared__ float Ss[64*64];
  __shared__ h16 Ps[64*64];
  const int tid = threadIdx.x, lane = tid & 31, wid = tid >> 5;
  const int win = blockIdx.x / NH_, head = blockIdx.x % NH_;
  const int wloc = win & 63, wi = wloc >> 3, wj = wloc & 7;
  for (int j = 0; j < 8; ++j){
    int idx = j*256 + tid;
    int row = idx >> 5, d = idx & 31;
    h16 q = (h16)0.f, kk = (h16)0.f, vv = (h16)0.f;
    if (row < 49){
      long base = ((long)win*49 + row)*1152 + head*HD_ + d;
      q  = (h16)((float)qkv[base] * 0.17677669529663689f);   // 1/sqrt(32)
      kk = qkv[base + 384];
      vv = qkv[base + 768];
    }
    Qs[row*32 + d] = q;
    Ks[row*32 + d] = kk;
    Vt[d*64 + row] = vv;
  }
  __syncthreads();
  const int kb = (lane & 16) ? 8 : 0, lr = lane & 15, rb = (lane & 16) ? 8 : 0;
  // S = Q K^T (64x64 padded), 16 tiles, 2 per wave
  for (int ti = 0; ti < 2; ++ti){
    int tile = wid*2 + ti, tr = tile >> 2, tc = tile & 3;
    Frag af = ldfrag(&Qs[(tr*16 + lr)*32], kb);
    Frag bf = ldfrag(&Ks[(tc*16 + lr)*32], kb);
    v8f acc = zero8();
    acc = __builtin_amdgcn_wmma_f32_16x16x32_f16(false, af.v, false, bf.v, (short)0, acc, false, false);
    int col = tc*16 + lr;
    for (int r = 0; r < 8; ++r) Ss[(tr*16 + rb + r)*64 + col] = acc[r];
  }
  __syncthreads();
  // bias + shift-mask + softmax, one thread per row
  if (tid < 64){
    int i = tid;
    if (i < 49){
      int pi = i/7, pj = i%7, gi = 0;
      if (shifted){
        int h2 = wi*7 + pi, w2 = wj*7 + pj;
        gi = (h2 > 52 ? 2 : (h2 > 48 ? 1 : 0))*3 + (w2 > 52 ? 2 : (w2 > 48 ? 1 : 0));
      }
      float mx = -1e30f;
      for (int jj = 0; jj < 49; ++jj){
        int qi = jj/7, qj = jj%7;
        float bs = rpb[((pi - qi + 6)*13 + (pj - qj + 6))*NH_ + head];
        if (shifted){
          int h2 = wi*7 + qi, w2 = wj*7 + qj;
          int gj = (h2 > 52 ? 2 : (h2 > 48 ? 1 : 0))*3 + (w2 > 52 ? 2 : (w2 > 48 ? 1 : 0));
          if (gj != gi) bs -= 100.f;
        }
        float s = Ss[i*64 + jj] + bs;
        Ss[i*64 + jj] = s;
        mx = fmaxf(mx, s);
      }
      float sum = 0.f;
      for (int jj = 0; jj < 49; ++jj){
        float e = __expf(Ss[i*64 + jj] - mx);
        Ss[i*64 + jj] = e; sum += e;
      }
      float inv = 1.f/sum;
      for (int jj = 0; jj < 49; ++jj) Ps[i*64 + jj] = (h16)(Ss[i*64 + jj]*inv);
      for (int jj = 49; jj < 64; ++jj) Ps[i*64 + jj] = (h16)0.f;
    } else {
      for (int jj = 0; jj < 64; ++jj) Ps[i*64 + jj] = (h16)0.f;
    }
  }
  __syncthreads();
  // O = P V : 64x32, 8 tiles, one per wave, K = 64 (2 steps)
  {
    int tr = wid & 3, tc = wid >> 2;
    v8f acc = zero8();
    for (int k0 = 0; k0 < 64; k0 += 32){
      Frag af = ldfrag(&Ps[(tr*16 + lr)*64 + k0], kb);
      Frag bf = ldfrag(&Vt[(tc*16 + lr)*64 + k0], kb);
      acc = __builtin_amdgcn_wmma_f32_16x16x32_f16(false, af.v, false, bf.v, (short)0, acc, false, false);
    }
    int d = tc*16 + lr;
    for (int r = 0; r < 8; ++r){
      int row = tr*16 + rb + r;
      if (row < 49)
        outp[((long)win*49 + row)*CC + head*HD_ + d] = (h16)acc[r];
    }
  }
}

// ---------------- LayerNorm / tails (wave per row) --------------------------
__global__ void k_ln1(const float* __restrict__ t, const float* __restrict__ g,
                      const float* __restrict__ be, h16* __restrict__ o, int shift){
  long row = (long)blockIdx.x*8 + (threadIdx.x >> 5);
  int lane = threadIdx.x & 31;
  if (row >= NTOK) return;
  const float* x = t + row*CC;
  float s = 0.f; for (int c = lane; c < CC; c += 32) s += x[c];
  float mu = wsum(s)/CC;
  float v = 0.f; for (int c = lane; c < CC; c += 32){ float d = x[c]-mu; v += d*d; }
  float rstd = rsqrtf(wsum(v)/CC + 1e-5f);
  int b = (int)(row / HW_), hw = (int)(row % HW_);
  int h2 = hw/56 - shift; if (h2 < 0) h2 += 56;
  int w2 = hw%56 - shift; if (w2 < 0) w2 += 56;
  long m = ((long)b*64 + (h2/7)*8 + (w2/7))*49 + (h2%7)*7 + (w2%7);
  h16* op = o + m*CC;
  for (int c = lane; c < CC; c += 32) op[c] = (h16)((x[c]-mu)*rstd*g[c] + be[c]);
}

__global__ void k_ln2(const float* __restrict__ t, const float* __restrict__ g,
                      const float* __restrict__ be, h16* __restrict__ o){
  long row = (long)blockIdx.x*8 + (threadIdx.x >> 5);
  int lane = threadIdx.x & 31;
  if (row >= NTOK) return;
  const float* x = t + row*CC;
  float s = 0.f; for (int c = lane; c < CC; c += 32) s += x[c];
  float mu = wsum(s)/CC;
  float v = 0.f; for (int c = lane; c < CC; c += 32){ float d = x[c]-mu; v += d*d; }
  float rstd = rsqrtf(wsum(v)/CC + 1e-5f);
  h16* op = o + row*CC;
  for (int c = lane; c < CC; c += 32) op[c] = (h16)((x[c]-mu)*rstd*g[c] + be[c]);
}

__global__ void k_pam_tail(const float* __restrict__ t, const float* __restrict__ x0,
                           const float* __restrict__ g, const float* __restrict__ be,
                           const float* __restrict__ gam, float* __restrict__ outPam,
                           h16* __restrict__ cat){
  long row = (long)blockIdx.x*8 + (threadIdx.x >> 5);
  int lane = threadIdx.x & 31;
  if (row >= NTOK) return;
  const float* x = t + row*CC;
  float s = 0.f; for (int c = lane; c < CC; c += 32) s += x[c];
  float mu = wsum(s)/CC;
  float v = 0.f; for (int c = lane; c < CC; c += 32){ float d = x[c]-mu; v += d*d; }
  float rstd = rsqrtf(wsum(v)/CC + 1e-5f);
  int b = (int)(row / HW_), hw = (int)(row % HW_);
  float gm = gam[0];
  for (int c = lane; c < CC; c += 32){
    float val = gm*((x[c]-mu)*rstd*g[c] + be[c]);
    long xi = ((long)b*CC + c)*HW_ + hw;
    float p = x0[xi] + val;
    outPam[xi] = p;
    cat[((long)b*768 + c)*HW_ + hw] = (h16)p;
  }
}

__global__ void k_cam_softmax(const float* __restrict__ E, h16* __restrict__ A){
  long row = (long)blockIdx.x*8 + (threadIdx.x >> 5);
  int lane = threadIdx.x & 31;
  if (row >= (long)BB*CC) return;
  const float* e = E + row*CC;
  float mn = 1e30f;
  for (int c = lane; c < CC; c += 32) mn = fminf(mn, e[c]);
  mn = wminf(mn);
  float s = 0.f;
  for (int c = lane; c < CC; c += 32) s += __expf(mn - e[c]);
  s = wsum(s);
  float inv = 1.f/s;
  h16* a = A + row*CC;
  for (int c = lane; c < CC; c += 32) a[c] = (h16)(__expf(mn - e[c])*inv);
}

__global__ void k_cam_tail(const float* __restrict__ camf, const float* __restrict__ x0,
                           const float* __restrict__ g, const float* __restrict__ be,
                           const float* __restrict__ gam, float* __restrict__ outCam,
                           h16* __restrict__ cat){
  int b = blockIdx.y;
  int n = blockIdx.x*256 + threadIdx.x;
  if (n >= HW_) return;
  const float* base = camf + (long)b*CC*HW_ + n;
  float mu = 0.f;
  for (int c = 0; c < CC; ++c) mu += base[(long)c*HW_];
  mu /= CC;
  float var = 0.f;
  for (int c = 0; c < CC; ++c){ float d = base[(long)c*HW_]-mu; var += d*d; }
  float rstd = rsqrtf(var/CC + 1e-5f);
  float gm = gam[0];
  for (int c = 0; c < CC; ++c){
    float v = gm*((base[(long)c*HW_]-mu)*rstd*g[c] + be[c]);
    long xi = ((long)b*CC + c)*HW_ + n;
    float o = x0[xi] + v;
    outCam[xi] = o;
    cat[((long)b*768 + 384 + c)*HW_ + n] = (h16)o;
  }
}

// ---------------- host launcher ---------------------------------------------
extern "C" void kernel_launch(void* const* d_in, const int* in_sizes, int n_in,
                              void* d_out_, int out_size, void* d_ws, size_t ws_size,
                              hipStream_t stream){
  (void)in_sizes; (void)n_in; (void)out_size; (void)ws_size;
  const float* x      = (const float*)d_in[0];
  const float* w_in   = (const float*)d_in[1];
  const float* s_in   = (const float*)d_in[2];
  const float* b_in   = (const float*)d_in[3];
  const float* ln1w   = (const float*)d_in[4];
  const float* ln1b   = (const float*)d_in[5];
  const float* qkvw   = (const float*)d_in[6];
  const float* qkvb   = (const float*)d_in[7];
  const float* projw  = (const float*)d_in[8];
  const float* projb  = (const float*)d_in[9];
  const float* rpb    = (const float*)d_in[10];
  const float* ln2w   = (const float*)d_in[11];
  const float* ln2b   = (const float*)d_in[12];
  const float* fc1w   = (const float*)d_in[13];
  const float* fc1b   = (const float*)d_in[14];
  const float* fc2w   = (const float*)d_in[15];
  const float* fc2b   = (const float*)d_in[16];
  const float* pamlnw = (const float*)d_in[17];
  const float* pamlnb = (const float*)d_in[18];
  const float* pamg   = (const float*)d_in[19];
  const float* camlnw = (const float*)d_in[20];
  const float* camlnb = (const float*)d_in[21];
  const float* camg   = (const float*)d_in[22];
  const float* w_out  = (const float*)d_in[23];
  const float* s_out  = (const float*)d_in[24];
  const float* b_out  = (const float*)d_in[25];
  float* out = (float*)d_out_;

  char* ws = (char*)d_ws;
  size_t off = 0;
  auto take = [&](size_t bytes)->char*{
    char* p = ws + off; off += (bytes + 255) & ~(size_t)255; return p;
  };
  h16*   w_in16  = (h16*)  take(884736ull*2);      // [384, 9*256]
  h16*   w_out16 = (h16*)  take(2654208ull*2);     // [384, 9*768]
  h16*   qkv16   = (h16*)  take(884736ull*2);      // 2 x [1152,384]
  h16*   proj16  = (h16*)  take(294912ull*2);      // 2 x [384,384]
  h16*   fc116   = (h16*)  take(1179648ull*2);     // 2 x [1536,384]
  h16*   fc216   = (h16*)  take(1179648ull*2);     // 2 x [384,1536]
  h16*   xin16   = (h16*)  take(12845056ull*2);    // input cast
  float* x0      = (float*)take((size_t)BCN*4);    // conv_in out
  h16*   x0h     = (h16*)  take((size_t)BCN*2);    // conv_in out f16
  float* t       = (float*)take((size_t)BCN*4);    // token stream (later camf)
  h16*   ybuf    = (h16*)  take((size_t)BCN*2);    // LN / attn-out f16
  h16*   big16   = (h16*)  take(77070336ull*2);    // qkv / mlp-hidden / concat
  float* energy  = (float*)take(2359296ull*4);     // [16,384,384]
  h16*   cattn   = (h16*)  take(2359296ull*2);

  // casts / reorders
  k_cast <<<2048, 256, 0, stream>>>(x, xin16, 12845056L);
  k_convw<<<1024, 256, 0, stream>>>(w_in, w_in16, 256);
  k_convw<<<2048, 256, 0, stream>>>(w_out, w_out16, 768);
  k_cast <<<1024, 256, 0, stream>>>(qkvw, qkv16, 884736L);
  k_cast <<<512,  256, 0, stream>>>(projw, proj16, 294912L);
  k_cast <<<1024, 256, 0, stream>>>(fc1w, fc116, 1179648L);
  k_cast <<<1024, 256, 0, stream>>>(fc2w, fc216, 1179648L);

  // conv_in + BN + GELU
  k_conv3x3<<<dim3(6, 784), 256, 0, stream>>>(xin16, w_in16, s_in, b_in, x0, x0h, 256);
  // [B,C,HW] -> [B,HW,C]
  k_transpose<<<8192, 256, 0, stream>>>(x0, t);

  // Swin blocks
  for (int i = 0; i < 2; ++i){
    int shift = i ? 3 : 0;
    k_ln1<<<6272, 256, 0, stream>>>(t, ln1w + i*CC, ln1b + i*CC, ybuf, shift);
    k_gemm_qkv<<<dim3(18, 392), 256, 0, stream>>>(
        ybuf, qkv16 + (long)i*442368, qkvb + i*1152, big16, 1152, 384);
    k_attn<<<12288, 256, 0, stream>>>(big16, rpb + (long)i*169*NH_, ybuf, i);
    k_gemm_proj<<<dim3(6, 392), 256, 0, stream>>>(
        ybuf, proj16 + (long)i*147456, projb + i*CC, t, 384, shift);
    k_ln2<<<6272, 256, 0, stream>>>(t, ln2w + i*CC, ln2b + i*CC, ybuf);
    k_gemm_fc1<<<dim3(24, 392), 256, 0, stream>>>(
        ybuf, fc116 + (long)i*589824, fc1b + i*MLP_, big16, MLP_, 384);
    k_gemm_fc2<<<dim3(6, 392), 256, 0, stream>>>(
        big16, fc216 + (long)i*589824, fc2b + i*CC, t, MLP_);
  }

  // PAM tail -> out slice 1 + concat channels [0,384)
  k_pam_tail<<<6272, 256, 0, stream>>>(t, x0, pamlnw, pamlnb, pamg, out + BCN, big16);

  // CAM
  k_gemm_energy <<<dim3(6, 3, 16),  256, 0, stream>>>(x0h, energy);
  k_cam_softmax <<<768, 256, 0, stream>>>(energy, cattn);
  k_gemm_camout <<<dim3(49, 6, 16), 256, 0, stream>>>(cattn, x0h, t);   // reuse t
  k_cam_tail    <<<dim3(13, 16), 256, 0, stream>>>(t, x0, camlnw, camlnb, camg,
                                                   out + 2*BCN, big16);

  // conv_out over concat -> out slice 0
  k_conv3x3<<<dim3(6, 784), 256, 0, stream>>>(big16, w_out16, s_out, b_out,
                                              out, (h16*)nullptr, 768);
}
